// _TorchKitNET_50216757625315
// MI455X (gfx1250) — compile-verified, matrix operand order checked by reading, not run on hardware
//
#include <hip/hip_runtime.h>
#include <math.h>

typedef __attribute__((ext_vector_type(16))) _Float16 v16h;
typedef __attribute__((ext_vector_type(8)))  float    v8f;
typedef __attribute__((ext_vector_type(4)))  float    f32x4;

#define B_ROWS 16384
#define F_DIM  2048
#define T_DIM  64
#define C_DIM  32
#define H_DIM  24
#define HH_DIM 48
#define GRIDX  32
#define WAVES_TOTAL (GRIDX * 8)          // 256 waves cooperating on one t
#define N_TILES (B_ROWS / 16)            // 1024 16-row tiles per t
#define N_ITER  (N_TILES / WAVES_TOTAL)  // exactly 4 tiles per wave
#define BUF_BYTES (8 * 16 * C_DIM * 4)   // one sX buffer: 16 KB

// ---------------------------------------------------------------------------
// Kernel 0: zero the per-t squared-error accumulators (harness poisons d_ws).
// ---------------------------------------------------------------------------
__global__ void zero_ws_kernel(float* __restrict__ ws) {
    if (threadIdx.x < T_DIM) ws[threadIdx.x] = 0.0f;
}

// 4 x 16B per lane: one wave's 16x32 f32 tile, global -> LDS via the async
// copy engine (ASYNCcnt-tracked, no VGPR round trip).
static __device__ __forceinline__ void async_copy_tile(uint32_t l0, uint32_t l2,
                                                       uint64_t g0, uint64_t g1) {
    asm volatile("global_load_async_to_lds_b128 %0, %1, off"
                 :: "v"(l0), "v"(g0) : "memory");
    asm volatile("global_load_async_to_lds_b128 %0, %1, off offset:16"
                 :: "v"(l0 + 16u), "v"(g0) : "memory");
    asm volatile("global_load_async_to_lds_b128 %0, %1, off"
                 :: "v"(l2), "v"(g1) : "memory");
    asm volatile("global_load_async_to_lds_b128 %0, %1, off offset:16"
                 :: "v"(l2 + 16u), "v"(g1) : "memory");
}

// ---------------------------------------------------------------------------
// Kernel 1: streaming pass. grid=(32,64): blockIdx.y = t, 8 waves/block,
// each wave owns exactly 4 independent 16-row tiles, double-buffered async
// staging with a fully unrolled (static-parity) pipeline.
// ---------------------------------------------------------------------------
__global__ __launch_bounds__(256) void tile_kernel(
    const float* __restrict__ x,        // (B, F)
    const int*   __restrict__ clusters, // (T, C)
    const float* __restrict__ Wt,       // (T, C, H)
    const float* __restrict__ hb,       // (T, H)
    const float* __restrict__ vb,       // (T, C)
    float*       __restrict__ ws)       // (T,) sum of squared errors
{
    __shared__ float    sW[C_DIM * H_DIM];                 // Wt[t] (c,h)
    __shared__ int      sCol[C_DIM];                       // clusters[t,:]
    __shared__ __align__(16) float sX[2][8][16][C_DIM];    // double-buffered xg tiles
    __shared__ __align__(16) float sH[8][16][C_DIM];       // per-wave h tile (f32,
                                                           //  cols 24..31 = exact 0)
    __shared__ float    sRed[8];
    __shared__ int      sContig;

    const int t      = blockIdx.y;
    const int tid    = threadIdx.x;
    const int lane   = tid & 31;
    const int wave   = tid >> 5;
    const int l16    = lane & 15;
    const int hiHalf = lane >> 4;

    if (tid == 0) sContig = 1;
    for (int i = tid; i < C_DIM * H_DIM; i += 256)
        sW[i] = Wt[t * C_DIM * H_DIM + i];
    if (tid < C_DIM) sCol[tid] = clusters[t * C_DIM + tid];
    __syncthreads();
    if (tid < C_DIM && sCol[tid] != sCol[0] + tid) atomicAnd(&sContig, 0);
    __syncthreads();
    const bool contig = (sContig != 0);      // uniform across block
    const int  col0   = sCol[0];

    // CDNA5 16-bit A-fragment K striping (wave32):
    //   v16h element i<8  -> K = klo + i ; i>=8 -> K = khi + (i-8)
    const int klo = hiHalf ? 8  : 0;
    const int khi = hiHalf ? 24 : 16;

    // LDS byte addresses for the async engine (buffer 0; buffer 1 = +BUF_BYTES).
    const uint32_t ldsRow = (uint32_t)(size_t)&sX[0][wave][l16][0];
    const uint32_t ldsW0  = ldsRow + klo * 4;
    const uint32_t ldsW2  = ldsRow + khi * 4;

    // Hot-path global addresses: two u64 regs bumped by a constant stride.
    const int waveGlobal = blockIdx.x * 8 + wave;       // 0..255
    const uint64_t gstride = (uint64_t)WAVES_TOTAL * 16ull * F_DIM * 4ull;
    uint64_t ga0 = 0, ga1 = 0;
    if (contig) {
        const size_t rowOff = (size_t)(waveGlobal * 16 + l16) * F_DIM;
        ga0 = (uint64_t)(size_t)x + (rowOff + (size_t)(col0 + klo)) * 4ull;
        ga1 = (uint64_t)(size_t)x + (rowOff + (size_t)(col0 + khi)) * 4ull;
        // Prefetch tile 0 into buffer 0; overlaps the fragment prologue below.
        async_copy_tile(ldsW0, ldsW2, ga0, ga1);
        ga0 += gstride;
        ga1 += gstride;
    }

    // ---- Loop-invariant B fragments (both GEMMs) from LDS ----
    v16h b1[2], b2[2];
#pragma unroll
    for (int nt = 0; nt < 2; ++nt) {
        const int n = nt * 16 + l16;
#pragma unroll
        for (int i = 0; i < 8; ++i) {
            const int k0 = klo + i, k1 = khi + i;
            b1[nt][i]     = (n  < H_DIM) ? (_Float16)sW[k0 * H_DIM + n] : (_Float16)0.f;
            b1[nt][i + 8] = (n  < H_DIM) ? (_Float16)sW[k1 * H_DIM + n] : (_Float16)0.f;
            b2[nt][i]     = (k0 < H_DIM) ? (_Float16)sW[n * H_DIM + k0] : (_Float16)0.f;
            b2[nt][i + 8] = (k1 < H_DIM) ? (_Float16)sW[n * H_DIM + k1] : (_Float16)0.f;
        }
    }

    int colA[16];                           // gather fallback columns
#pragma unroll
    for (int i = 0; i < 8; ++i) { colA[i] = sCol[klo + i]; colA[i + 8] = sCol[khi + i]; }

    const float hbv0 = hb[t * H_DIM + l16];
    // hbv1 = 0 for l16 >= 8: together with the zero B1 columns this makes the
    // unconditional spill below write exact zeros into sH cols 24..31.
    const float hbv1 = (l16 < H_DIM - 16) ? hb[t * H_DIM + 16 + l16] : 0.f;
    const float vbv0 = vb[t * C_DIM + l16];
    const float vbv1 = vb[t * C_DIM + 16 + l16];

    const v8f zero_acc = {};
    float acc = 0.f;

#pragma unroll
    for (int it = 0; it < N_ITER; ++it) {
        const int parity = it & 1;                    // compile-time after unroll
        const int tile   = waveGlobal + it * WAVES_TOTAL;

        if (contig) {
            if (it + 1 < N_ITER) {
                // Outstanding DS reads of the other buffer must land before the
                // (DS-unordered) async engine overwrites it.
                asm volatile("s_wait_dscnt 0x0" ::: "memory");
                const uint32_t nb = (parity ^ 1) * (uint32_t)BUF_BYTES;
                async_copy_tile(ldsW0 + nb, ldsW2 + nb, ga0, ga1);
                ga0 += gstride;
                ga1 += gstride;
                // Wait only for the CURRENT buffer's 4 copies (async loads
                // complete in order): <=4 outstanding leaves the new batch flying.
                asm volatile("s_wait_asynccnt 0x4" ::: "memory");
            } else {
                asm volatile("s_wait_asynccnt 0x0" ::: "memory");
            }
        } else {
            const float* xrow = x + (size_t)(tile * 16 + l16) * F_DIM;
            float* srow = &sX[parity][wave][l16][0];
#pragma unroll
            for (int i = 0; i < 8; ++i) {
                srow[klo + i] = xrow[colA[i]];
                srow[khi + i] = xrow[colA[i + 8]];
            }
        }

        // ---- A fragment: 4 x ds_load_b128 + packed cvt ----
        const f32x4* ap = (const f32x4*)&sX[parity][wave][l16][0];
        const f32x4 q0 = ap[klo / 4], q1 = ap[klo / 4 + 1];
        const f32x4 q2 = ap[khi / 4], q3 = ap[khi / 4 + 1];
        v16h a;
#pragma unroll
        for (int i = 0; i < 4; ++i) {
            a[i]      = (_Float16)q0[i];
            a[i + 4]  = (_Float16)q1[i];
            a[i + 8]  = (_Float16)q2[i];
            a[i + 12] = (_Float16)q3[i];
        }

        // ---- GEMM1: h = xg @ Wt ----
        v8f c1a = __builtin_amdgcn_wmma_f32_16x16x32_f16(false, a, false, b1[0],
                                                         (short)0, zero_acc, false, false);
        v8f c1b = __builtin_amdgcn_wmma_f32_16x16x32_f16(false, a, false, b1[1],
                                                         (short)0, zero_acc, false, false);

        // h (+hb) spill, f32, unconditional: both stores per row share one base
        // (offsets r*128 and r*128+64) -> ds_store_2addr_b32, no exec toggling.
        {
            float* hrow = &sH[wave][8 * hiHalf][0] + l16;
#pragma unroll
            for (int r = 0; r < 8; ++r) {
                hrow[r * C_DIM]      = c1a[r] + hbv0;
                hrow[r * C_DIM + 16] = c1b[r] + hbv1;   // exact 0 for cols >= 24
            }
        }

        // ---- A2 fragment: 4 x ds_load_b128 (cols 24..31 already zero) ----
        const f32x4* hp = (const f32x4*)&sH[wave][l16][0];
        const f32x4 h0 = hp[klo / 4], h1 = hp[klo / 4 + 1];
        const f32x4 h2 = hp[khi / 4], h3 = hp[khi / 4 + 1];
        v16h a2;
#pragma unroll
        for (int i = 0; i < 4; ++i) {
            a2[i]      = (_Float16)h0[i];
            a2[i + 4]  = (_Float16)h1[i];
            a2[i + 8]  = (_Float16)h2[i];
            a2[i + 12] = (_Float16)h3[i];
        }

        // ---- GEMM2: out = h @ Wt^T ----
        v8f c2a = __builtin_amdgcn_wmma_f32_16x16x32_f16(false, a2, false, b2[0],
                                                         (short)0, zero_acc, false, false);
        v8f c2b = __builtin_amdgcn_wmma_f32_16x16x32_f16(false, a2, false, b2[1],
                                                         (short)0, zero_acc, false, false);

        // ---- Residual^2 from the LDS f32 tile (immediate-offset ds reads) ----
        const float* dbase = &sX[parity][wave][8 * hiHalf][0] + l16;
#pragma unroll
        for (int r = 0; r < 8; ++r) {
            const float xg0 = dbase[r * C_DIM];
            const float xg1 = dbase[r * C_DIM + 16];
            const float d0 = (c2a[r] + vbv0) - xg0;
            const float d1 = (c2b[r] + vbv1) - xg1;
            acc = fmaf(d0, d0, acc);
            acc = fmaf(d1, d1, acc);
        }
    }

    // ---- Reduce: wave shuffle -> LDS -> one atomicAdd per block ----
#pragma unroll
    for (int off = 16; off > 0; off >>= 1)
        acc += __shfl_down(acc, off, 32);
    if (lane == 0) sRed[wave] = acc;
    __syncthreads();
    if (tid == 0) {
        float s = 0.f;
#pragma unroll
        for (int w = 0; w < 8; ++w) s += sRed[w];
        atomicAdd(&ws[t], s);
    }
}

// ---------------------------------------------------------------------------
// Kernel 2: rmse -> tails -> tiny head MLP. One block of 64 threads.
// d_out[0:64] = head_out, d_out[64:128] = tails.
// ---------------------------------------------------------------------------
__global__ void head_kernel(const float* __restrict__ ws,
                            const float* __restrict__ Wh,   // (T, HH)
                            const float* __restrict__ hbh,  // (HH,)
                            const float* __restrict__ vbh,  // (T,)
                            float* __restrict__ out)
{
    __shared__ float sT[T_DIM];
    __shared__ float sHH[HH_DIM];
    const int tid = threadIdx.x;

    if (tid < T_DIM) {
        const float rmse = sqrtf(ws[tid] / (float)((size_t)B_ROWS * C_DIM));
        const float tail = logf(rmse);
        sT[tid] = tail;
        out[T_DIM + tid] = tail;
    }
    __syncthreads();
    if (tid < HH_DIM) {
        float s = hbh[tid];
        for (int i = 0; i < T_DIM; ++i) s += sT[i] * Wh[i * HH_DIM + tid];
        sHH[tid] = s;
    }
    __syncthreads();
    if (tid < T_DIM) {
        float s = vbh[tid];
        for (int j = 0; j < HH_DIM; ++j) s += sHH[j] * Wh[tid * HH_DIM + j];
        out[tid] = s;
    }
}

// ---------------------------------------------------------------------------
extern "C" void kernel_launch(void* const* d_in, const int* in_sizes, int n_in,
                              void* d_out, int out_size, void* d_ws, size_t ws_size,
                              hipStream_t stream) {
    const float* x        = (const float*)d_in[0];
    const int*   clusters = (const int*)  d_in[1];
    const float* Wt       = (const float*)d_in[2];
    const float* hb       = (const float*)d_in[3];
    const float* vb       = (const float*)d_in[4];
    const float* Wh       = (const float*)d_in[5];
    const float* hbh      = (const float*)d_in[6];
    const float* vbh      = (const float*)d_in[7];
    float* out = (float*)d_out;
    float* ws  = (float*)d_ws;   // first 64 floats = per-t sum of squared err

    zero_ws_kernel<<<1, 64, 0, stream>>>(ws);
    dim3 grid(GRIDX, T_DIM, 1);
    tile_kernel<<<grid, 256, 0, stream>>>(x, clusters, Wt, hb, vb, ws);
    head_kernel<<<1, 64, 0, stream>>>(ws, Wh, hbh, vbh, out);
}
